// Attention_18056042512720
// MI455X (gfx1250) — compile-verified, matrix operand order checked by reading, not run on hardware
//
#include <hip/hip_runtime.h>
#include <hip/hip_bf16.h>

// ---------------------------------------------------------------------------
// Types / helpers
// ---------------------------------------------------------------------------
typedef __attribute__((ext_vector_type(16))) __bf16 v16bf;
typedef __attribute__((ext_vector_type(8)))  float  v8f;
typedef __attribute__((ext_vector_type(4)))  unsigned int u32x4;
typedef __attribute__((ext_vector_type(8)))  unsigned int u32x8;

union Frag16 {           // 16 bf16 = 32 bytes = two b128 chunks
  u32x4 u[2];
  v16bf v;
};

__device__ __forceinline__ unsigned short f2bf(float f) {
  unsigned int u = __float_as_uint(f);
  u += 0x7FFFu + ((u >> 16) & 1u);   // round-to-nearest-even
  return (unsigned short)(u >> 16);
}

#define ATT_L     2048
#define ATT_HID   2048
#define ATT_NH    32
#define ATT_D     64
#define ATT_PAD   1792          // keys >= PAD are padding-masked
#define ATT_SCALE 0.125f
#define ATT_NEG   (-1.0e9f)

// ---------------------------------------------------------------------------
// Tiled GEMM + bias:  C[M,N] = A[M,K] * W[K,N] + bias[N]
// Block tile 64x128, BK=32, 8 waves (4x2), each wave 16x64 via 4 WMMA accs.
// A may be f32 (converted to bf16 on the fly) or bf16; C may be bf16 or f32.
// ---------------------------------------------------------------------------
template <bool A_IS_BF16, bool OUT_BF16>
__global__ __launch_bounds__(256)
void gemm_bias_kernel(const void* __restrict__ Aptr,
                      const float* __restrict__ W,
                      const float* __restrict__ bias,
                      void* __restrict__ Cptr,
                      int M, int N, int K)
{
  constexpr int BM = 64, BN = 128, BK = 32;
  __shared__ __align__(16) unsigned short sA[BM * BK];   // [m][k]
  __shared__ __align__(16) unsigned short sB[BN * BK];   // [n][k] (transposed)

  const int tid  = threadIdx.x;
  const int lane = tid & 31;
  const int wave = tid >> 5;
  const int wm   = wave >> 1;       // 0..3 : 16-row slice
  const int wn   = wave & 1;        // 0..1 : 64-col slice
  const int rl   = lane & 15;
  const int half = lane >> 4;       // 0/1 : which K-half of the fragment
  const int m0   = blockIdx.x * BM;
  const int n0   = blockIdx.y * BN;

  const float*          Af = (const float*)Aptr;
  const unsigned short* Ab = (const unsigned short*)Aptr;

  v8f acc[4] = {};

  for (int kt = 0; kt < K; kt += BK) {
    if (kt + BK < K)  // warm next W tile into cache (global_prefetch_b8)
      __builtin_prefetch(&W[(size_t)(kt + BK) * N + n0 + tid], 0, 1);

    // ---- stage A tile (64x32) ----
#pragma unroll
    for (int i = 0; i < (BM * BK) / 256; ++i) {
      int idx = tid + i * 256;
      int m = idx >> 5, k = idx & 31;
      unsigned short ab;
      if constexpr (A_IS_BF16) ab = Ab[(size_t)(m0 + m) * K + kt + k];
      else                     ab = f2bf(Af[(size_t)(m0 + m) * K + kt + k]);
      sA[idx] = ab;
    }
    // ---- stage B tile transposed: sB[n][k] ----
#pragma unroll
    for (int i = 0; i < (BN * BK) / 256; ++i) {
      int idx = tid + i * 256;
      int k = idx >> 7;              // /BN
      int n = idx & 127;             // %BN
      sB[n * BK + k] = f2bf(W[(size_t)(kt + k) * N + n0 + n]);
    }
    __syncthreads();

    // ---- A fragment: lane rl -> row, halves hold K {0..7,16..23}/{8..15,24..31}
    Frag16 a;
    const unsigned short* pa = &sA[(wm * 16 + rl) * BK + half * 8];
    a.u[0] = *(const u32x4*)(pa);
    a.u[1] = *(const u32x4*)(pa + 16);

#pragma unroll
    for (int t = 0; t < 4; ++t) {
      Frag16 bfr;
      const unsigned short* pb = &sB[(wn * 64 + t * 16 + rl) * BK + half * 8];
      bfr.u[0] = *(const u32x4*)(pb);
      bfr.u[1] = *(const u32x4*)(pb + 16);
      acc[t] = __builtin_amdgcn_wmma_f32_16x16x32_bf16(
          false, a.v, false, bfr.v, (short)0, acc[t], false, false);
    }
    __syncthreads();
  }

  // ---- epilogue: C layout = row (vgpr, +8 for lanes 16..31), col (lane) ----
#pragma unroll
  for (int t = 0; t < 4; ++t) {
    const int col = n0 + wn * 64 + t * 16 + rl;
    const float bv = bias[col];
#pragma unroll
    for (int r = 0; r < 8; ++r) {
      const int row = m0 + wm * 16 + half * 8 + r;
      const float v = acc[t][r] + bv;
      if constexpr (OUT_BF16)
        ((unsigned short*)Cptr)[(size_t)row * N + col] = f2bf(v);
      else
        ((float*)Cptr)[(size_t)row * N + col] = v;
    }
  }
}

// ---------------------------------------------------------------------------
// Flash attention (GQA): grid (L/64, NH, B), 128 threads = 4 waves.
// Each wave owns 16 q rows. K tile is fetched by the Tensor Data Mover
// (tensor_load_to_lds, TENSORcnt); V tile is manually transposed into LDS.
// Causal + padding masks are analytic; key loop stops at min(q0+64, PAD).
// ---------------------------------------------------------------------------
__global__ __launch_bounds__(128)
void attn_kernel(const unsigned short* __restrict__ qp,    // [B*L, 2048] bf16
                 const unsigned short* __restrict__ kvp,   // [B*L, 1024] bf16
                 unsigned short* __restrict__ op)          // [B*L, 2048] bf16
{
  __shared__ __align__(16) unsigned short sK[32 * 64];       // [key][d]
  __shared__ __align__(16) unsigned short sV[64 * 32];       // [d][key] (transposed)
  __shared__ __align__(16) unsigned short sP[4][16 * 32];    // per-wave P scratch

  const int tid  = threadIdx.x;
  const int lane = tid & 31;
  const int wave = tid >> 5;
  const int rl   = lane & 15;
  const int half = lane >> 4;

  const int q0  = blockIdx.x * 64;
  const int h   = blockIdx.y;
  const int b   = blockIdx.z;
  const int kvh = h >> 2;              // repeat_interleave(4)
  const int qw  = q0 + wave * 16;

  // Q fragments straight from global in WMMA A-layout (contiguous 16B chunks)
  const size_t qrow = ((size_t)(b * ATT_L + qw + rl)) * ATT_HID + h * ATT_D;
  Frag16 qf[2];
#pragma unroll
  for (int c = 0; c < 2; ++c) {
    qf[c].u[0] = *(const u32x4*)(qp + qrow + c * 32 + half * 8);
    qf[c].u[1] = *(const u32x4*)(qp + qrow + c * 32 + half * 8 + 16);
  }

  v8f o[4] = {};
  float mrun[8], lrun[8];
#pragma unroll
  for (int r = 0; r < 8; ++r) { mrun[r] = -3.0e38f; lrun[r] = 0.f; }

  const int    kmax   = min(q0 + 64, ATT_PAD);
  const size_t kvbase = (size_t)b * ATT_L * 1024;

  // ---- TDM descriptor group 1 (loop-invariant): 2-byte elems, 32x64 tile,
  //      tensor_dim0=64, tensor_dim1=32, row stride = 1024 elems ----
  u32x8 g1;
  g1[0] = 0x00010000u;     // [15:0] wg_mask=0 (not in cluster), [17:16] data_size=1 (2B)
  g1[1] = (64u << 16);     // tensor_dim0[15:0]=64  (bits 79:48 of group1)
  g1[2] = (32u << 16);     // tensor_dim1[15:0]=32  (bits 111:80)
  g1[3] = (64u << 16);     // tile_dim0=64          (bits 127:112)
  g1[4] = 32u;             // tile_dim1=32          (bits 143:128), tile_dim2=0
  g1[5] = 1024u;           // tensor_dim0_stride[31:0]=1024 (bits 207:160)
  g1[6] = 0u;              // stride hi / tensor_dim1_stride = 0
  g1[7] = 0u;
  const unsigned ldsK = (unsigned)(size_t)&sK[0];  // generic addr[31:0] == LDS offset

  for (int k0 = 0; k0 < kmax; k0 += 32) {
    // ---- K tile via Tensor Data Mover (one issue per workgroup) ----
    if (wave == 0) {
      const unsigned long long ga =
          (unsigned long long)(const void*)(kvp + kvbase + (size_t)k0 * 1024 + kvh * 64);
      u32x4 g0;
      g0[0] = 1u;                                    // count=1, user mode
      g0[1] = ldsK;                                  // lds_addr (bytes)
      g0[2] = (unsigned)(ga & 0xffffffffu);          // global_addr[31:0]
      g0[3] = (unsigned)((ga >> 32) & 0x01ffffffu)   // global_addr[56:32]
              | (2u << 30);                          // type = 2 ("image")
      asm volatile("tensor_load_to_lds %0, %1" :: "s"(g0), "s"(g1) : "memory");
    }

    // ---- stage V transposed: sV[d][key] (overlaps with the TDM transfer) ----
    for (int i = tid; i < 2048; i += 128) {
      int key = i >> 6, d = i & 63;
      sV[d * 32 + key] = kvp[kvbase + (size_t)(k0 + key) * 1024 + 512 + kvh * 64 + d];
    }
    if (wave == 0) __builtin_amdgcn_s_wait_tensorcnt(0);
    __syncthreads();

    // ---- S = Q * K^T : 2 key-tiles x 2 K-dim chunks = 4 WMMA ----
    v8f s[2] = {};
#pragma unroll
    for (int t = 0; t < 2; ++t) {
#pragma unroll
      for (int c = 0; c < 2; ++c) {
        Frag16 kf;
        const unsigned short* pk = &sK[(t * 16 + rl) * 64 + c * 32 + half * 8];
        kf.u[0] = *(const u32x4*)pk;
        kf.u[1] = *(const u32x4*)(pk + 16);
        s[t] = __builtin_amdgcn_wmma_f32_16x16x32_bf16(
            false, qf[c].v, false, kf.v, (short)0, s[t], false, false);
      }
    }

    // ---- scale + causal mask + online softmax (rows live across 16 lanes) ----
#pragma unroll
    for (int r = 0; r < 8; ++r) {
      const int row = qw + half * 8 + r;
      float x0 = s[0][r] * ATT_SCALE;
      float x1 = s[1][r] * ATT_SCALE;
      if (k0 + rl      > row) x0 = ATT_NEG;
      if (k0 + 16 + rl > row) x1 = ATT_NEG;
      float mx = fmaxf(x0, x1);
#pragma unroll
      for (int off = 1; off < 16; off <<= 1)
        mx = fmaxf(mx, __shfl_xor(mx, off, 32));
      const float mnew  = fmaxf(mrun[r], mx);
      const float alpha = __expf(mrun[r] - mnew);
      const float p0 = __expf(x0 - mnew);
      const float p1 = __expf(x1 - mnew);
      float ps = p0 + p1;
#pragma unroll
      for (int off = 1; off < 16; off <<= 1)
        ps += __shfl_xor(ps, off, 32);
      lrun[r] = lrun[r] * alpha + ps;
      mrun[r] = mnew;
#pragma unroll
      for (int t4 = 0; t4 < 4; ++t4) o[t4][r] *= alpha;
      // C-layout -> LDS (row-major 16x32) for A-layout reload
      sP[wave][(half * 8 + r) * 32 + rl]      = f2bf(p0);
      sP[wave][(half * 8 + r) * 32 + 16 + rl] = f2bf(p1);
    }

    // ---- O += P * V : per-wave LDS (DS ops are in-order within a wave) ----
    Frag16 pf;
    const unsigned short* pp = &sP[wave][rl * 32 + half * 8];
    pf.u[0] = *(const u32x4*)pp;
    pf.u[1] = *(const u32x4*)(pp + 16);
#pragma unroll
    for (int t4 = 0; t4 < 4; ++t4) {
      Frag16 vf;
      const unsigned short* pv = &sV[(t4 * 16 + rl) * 32 + half * 8];
      vf.u[0] = *(const u32x4*)pv;
      vf.u[1] = *(const u32x4*)(pv + 16);
      o[t4] = __builtin_amdgcn_wmma_f32_16x16x32_bf16(
          false, pf.v, false, vf.v, (short)0, o[t4], false, false);
    }
    __syncthreads();
  }

  // ---- normalize and store attn output (bf16, [B,L,H*D]) ----
#pragma unroll
  for (int t4 = 0; t4 < 4; ++t4) {
#pragma unroll
    for (int r = 0; r < 8; ++r) {
      const int row = qw + half * 8 + r;
      const float v = o[t4][r] / lrun[r];
      op[((size_t)(b * ATT_L + row)) * ATT_HID + h * ATT_D + t4 * 16 + rl] = f2bf(v);
    }
  }
}

// ---------------------------------------------------------------------------
// Launch
// ---------------------------------------------------------------------------
extern "C" void kernel_launch(void* const* d_in, const int* in_sizes, int n_in,
                              void* d_out, int out_size, void* d_ws, size_t ws_size,
                              hipStream_t stream) {
  const float* query = (const float*)d_in[0];
  const float* kv    = (const float*)d_in[1];
  const float* Wq    = (const float*)d_in[2];
  const float* bq    = (const float*)d_in[3];
  const float* Wkv   = (const float*)d_in[4];
  const float* bkv   = (const float*)d_in[5];
  const float* Wo    = (const float*)d_in[6];
  const float* bo    = (const float*)d_in[7];
  float* out = (float*)d_out;

  const int Mrows = 2 * ATT_L;  // B*L = 4096

  unsigned short* qpb  = (unsigned short*)d_ws;                 // 4096*2048 bf16
  unsigned short* kvpb = qpb  + (size_t)Mrows * ATT_HID;        // 4096*1024 bf16
  unsigned short* attn = kvpb + (size_t)Mrows * 1024;           // 4096*2048 bf16

  // Q projection: f32 A -> bf16 out
  gemm_bias_kernel<false, true><<<dim3(Mrows / 64, ATT_HID / 128), 256, 0, stream>>>(
      query, Wq, bq, qpb, Mrows, ATT_HID, ATT_HID);
  // KV projection: f32 A -> bf16 out (N = 1024)
  gemm_bias_kernel<false, true><<<dim3(Mrows / 64, 1024 / 128), 256, 0, stream>>>(
      kv, Wkv, bkv, kvpb, Mrows, 1024, ATT_HID);
  // Flash attention
  attn_kernel<<<dim3(ATT_L / 64, ATT_NH, 2), 128, 0, stream>>>(qpb, kvpb, attn);
  // Output projection: bf16 A -> f32 out
  gemm_bias_kernel<true, false><<<dim3(Mrows / 64, ATT_HID / 128), 256, 0, stream>>>(
      attn, Wo, bo, out, Mrows, ATT_HID, ATT_HID);
}
